// GConvGRUModel_79585743995076
// MI455X (gfx1250) — matrix operand order repaired and checked by reading, not compiled, not run on hardware
//
#include <hip/hip_runtime.h>
#include <math.h>

// GConvGRU step for MI455X (gfx1250, wave32).
// Node arrays (12.8MB) + edges (~19MB) are L2-resident (192MB L2), so the
// scatter SpMMs are L2-atomic-throughput bound: wave-per-edge, lane=channel,
// no-return relaxed f32 atomics. Dense weight application is done with
// v_wmma_f32_16x16x32_f16 on 16-node tiles, f32 accumulation, weights
// pre-converted once to f16 in WMMA-B register order.

typedef __attribute__((ext_vector_type(16))) _Float16 v16h;
typedef __attribute__((ext_vector_type(8)))  float    v8f;
typedef __attribute__((ext_vector_type(4)))  float    v4f;

#define CCH 32

// No-return relaxed device-scope f32 atomic add -> global_atomic_add_f32 (no RTN).
__device__ __forceinline__ void atomAddNoRet(float* p, float v) {
  (void)__hip_atomic_fetch_add(p, v, __ATOMIC_RELAXED, __HIP_MEMORY_SCOPE_AGENT);
}

// ---------------------------------------------------------------- utilities
// n must be a multiple of 4 and p 16B-aligned (all our regions are).
__global__ void zero_f32x4(v4f* p, long n4) {
  long i = (long)blockIdx.x * blockDim.x + threadIdx.x;
  long stride = (long)gridDim.x * blockDim.x;
  v4f z = {0.f, 0.f, 0.f, 0.f};
  for (; i < n4; i += stride) p[i] = z;
}

// Convert fp32 weights to f16 in WMMA-B layout.
// Wzr: [kk 0..3][k 0..31][col 0..63]  col<32 -> gate z, else gate r
//   kk: 0->Wx[g][0], 1->Wx[g][1], 2->Wh[g][0], 3->Wh[g][1]
// Wc:  [kk][k][j] for gate 2 (candidate state)
__global__ void cvt_weights(const float* __restrict__ Wx,
                            const float* __restrict__ Wh,
                            _Float16* __restrict__ Wzr,
                            _Float16* __restrict__ Wc) {
  int i = blockIdx.x * blockDim.x + threadIdx.x;
  if (i < 4 * 32 * 64) {
    int col = i & 63;
    int k   = (i >> 6) & 31;
    int kk  = i >> 11;
    int g = col >> 5, j = col & 31;
    const float* W = (kk < 2) ? Wx : Wh;
    int s = kk & 1;
    Wzr[i] = (_Float16)W[((g * 2 + s) * 32 + k) * 32 + j];
  } else {
    int i2 = i - 4 * 32 * 64;
    if (i2 < 4 * 32 * 32) {
      int j  = i2 & 31;
      int k  = (i2 >> 5) & 31;
      int kk = i2 >> 10;
      const float* W = (kk < 2) ? Wx : Wh;
      int s = kk & 1;
      Wc[i2] = (_Float16)W[((2 * 2 + s) * 32 + k) * 32 + j];
    }
  }
}

// ---------------------------------------------------------------- graph part
__global__ void k_degree(const int* __restrict__ src, const float* __restrict__ w,
                         float* __restrict__ deg, int E) {
  int e = blockIdx.x * blockDim.x + threadIdx.x;
  if (e < E) atomAddNoRet(deg + src[e], w[e]);
}

__global__ void k_coef(const int* __restrict__ src, const int* __restrict__ dst,
                       const float* __restrict__ w, const float* __restrict__ deg,
                       float* __restrict__ coef, int E) {
  int e = blockIdx.x * blockDim.x + threadIdx.x;
  if (e >= E) return;
  float ds = deg[src[e]], dd = deg[dst[e]];
  float is = ds > 0.f ? rsqrtf(ds) : 0.f;
  float id = dd > 0.f ? rsqrtf(dd) : 0.f;
  coef[e] = -(is * w[e] * id);
}

// Wave-per-edge SpMM: lane = channel. Fused for x and h (same edge walk).
__global__ void k_spmm2(const int* __restrict__ src, const int* __restrict__ dst,
                        const float* __restrict__ coef,
                        const float* __restrict__ x, const float* __restrict__ h,
                        float* __restrict__ tx, float* __restrict__ th, int E) {
  int lane = threadIdx.x & 31;
  int e = blockIdx.x * (blockDim.x >> 5) + (threadIdx.x >> 5);
  if (e >= E) return;
  int s = src[e], d = dst[e];
  float c = coef[e];
  atomAddNoRet(tx + (size_t)d * CCH + lane, c * x[(size_t)s * CCH + lane]);
  atomAddNoRet(th + (size_t)d * CCH + lane, c * h[(size_t)s * CCH + lane]);
}

__global__ void k_spmm1(const int* __restrict__ src, const int* __restrict__ dst,
                        const float* __restrict__ coef,
                        const float* __restrict__ v, float* __restrict__ tv, int E) {
  int lane = threadIdx.x & 31;
  int e = blockIdx.x * (blockDim.x >> 5) + (threadIdx.x >> 5);
  if (e >= E) return;
  int s = src[e], d = dst[e];
  float c = coef[e];
  atomAddNoRet(tv + (size_t)d * CCH + lane, c * v[(size_t)s * CCH + lane]);
}

// ----------------------------------------------------------------- WMMA part
// A-operand (16x32 f16) per CDNA5 layout: lane M = lane&15,
// components 0..7  -> K = 8*(lane>>4) + 0..7
// components 8..15 -> K = 16 + 8*(lane>>4) + 0..7
__device__ __forceinline__ v16h load_a_f16(const float* __restrict__ seg,
                                           int node, int lane) {
  const float* rp = seg + (size_t)node * CCH;
  int bk = (lane >> 4) * 8;
  v4f p0 = *(const v4f*)(rp + bk);
  v4f p1 = *(const v4f*)(rp + bk + 4);
  v4f q0 = *(const v4f*)(rp + 16 + bk);
  v4f q1 = *(const v4f*)(rp + 16 + bk + 4);
  v16h a;
#pragma unroll
  for (int i = 0; i < 4; ++i) {
    a[i]      = (_Float16)p0[i];
    a[4 + i]  = (_Float16)p1[i];
    a[8 + i]  = (_Float16)q0[i];
    a[12 + i] = (_Float16)q1[i];
  }
  return a;
}

// Gates z and r fused: U = [x | Ax | h | Ah] (K=128) against Wzr (N=64: z|r).
__global__ void k_gates_zr(const float* __restrict__ x, const float* __restrict__ tx,
                           const float* __restrict__ h, const float* __restrict__ th,
                           const _Float16* __restrict__ Wzr,
                           const float* __restrict__ bx, const float* __restrict__ bh,
                           float* __restrict__ zbuf, float* __restrict__ hrbuf, int n) {
  int lane = threadIdx.x & 31;
  int tile = blockIdx.x * (blockDim.x >> 5) + (threadIdx.x >> 5);
  int ntiles = (n + 15) >> 4;
  if (tile >= ntiles) return;          // wave-uniform: EXEC stays all-1 for WMMA
  int nodeA = tile * 16 + (lane & 15);
  int nodeL = nodeA < n ? nodeA : n - 1;
  const float* segs[4] = {x, tx, h, th};

  v8f acc[4] = {};
#pragma unroll
  for (int kk = 0; kk < 4; ++kk) {
    v16h a = load_a_f16(segs[kk], nodeL, lane);
    const _Float16* bbase = Wzr + (kk * 32 * 64) + (lane & 15);  // lane&15 = N column
#pragma unroll
    for (int nt = 0; nt < 4; ++nt) {
      v16h b;
#pragma unroll
      for (int s = 0; s < 16; ++s)
        b[s] = bbase[((lane >> 4) * 8 + (s & 7) + (s >> 3) * 16) * 64 + nt * 16];  // table-order K
      acc[nt] = __builtin_amdgcn_wmma_f32_16x16x32_f16(
          false, a, false, b, (short)0, acc[nt], false, false);
    }
  }

  int half = lane >> 4;
  int cidx = lane & 15;
#pragma unroll
  for (int nt = 0; nt < 4; ++nt) {
    int g = nt >> 1;                    // 0 -> z, 1 -> r
    int j = (nt & 1) * 16 + cidx;
    float bias = bx[g * CCH + j] + bh[g * CCH + j];
#pragma unroll
    for (int r = 0; r < 8; ++r) {
      int node = tile * 16 + half * 8 + r;   // C layout: M = r + 8*(lane>>4)
      if (node >= n) continue;
      float v = acc[nt][r] + bias;
      float sg = 1.f / (1.f + __expf(-v));
      if (g == 0) zbuf[(size_t)node * CCH + j] = sg;
      else        hrbuf[(size_t)node * CCH + j] = sg * h[(size_t)node * CCH + j];
    }
  }
}

// Candidate gate: U2 = [x | Ax | h*r | A(h*r)] vs Wc (N=32), then h_new.
__global__ void k_gate_c(const float* __restrict__ x, const float* __restrict__ tx,
                         const float* __restrict__ hr, const float* __restrict__ thr,
                         const _Float16* __restrict__ Wc,
                         const float* __restrict__ bx, const float* __restrict__ bh,
                         const float* __restrict__ h, const float* __restrict__ zbuf,
                         float* __restrict__ hnew, int n) {
  int lane = threadIdx.x & 31;
  int tile = blockIdx.x * (blockDim.x >> 5) + (threadIdx.x >> 5);
  int ntiles = (n + 15) >> 4;
  if (tile >= ntiles) return;
  int nodeA = tile * 16 + (lane & 15);
  int nodeL = nodeA < n ? nodeA : n - 1;
  const float* segs[4] = {x, tx, hr, thr};

  v8f acc[2] = {};
#pragma unroll
  for (int kk = 0; kk < 4; ++kk) {
    v16h a = load_a_f16(segs[kk], nodeL, lane);
    const _Float16* bbase = Wc + (kk * 32 * 32) + (lane & 15);
#pragma unroll
    for (int nt = 0; nt < 2; ++nt) {
      v16h b;
#pragma unroll
      for (int s = 0; s < 16; ++s)
        b[s] = bbase[((lane >> 4) * 8 + (s & 7) + (s >> 3) * 16) * 32 + nt * 16];
      acc[nt] = __builtin_amdgcn_wmma_f32_16x16x32_f16(
          false, a, false, b, (short)0, acc[nt], false, false);
    }
  }

  int half = lane >> 4;
  int cidx = lane & 15;
#pragma unroll
  for (int nt = 0; nt < 2; ++nt) {
    int j = nt * 16 + cidx;
    float bias = bx[2 * CCH + j] + bh[2 * CCH + j];
#pragma unroll
    for (int r = 0; r < 8; ++r) {
      int node = tile * 16 + half * 8 + r;
      if (node >= n) continue;
      float ht = tanhf(acc[nt][r] + bias);
      float zv = zbuf[(size_t)node * CCH + j];
      float hv = h[(size_t)node * CCH + j];
      hnew[(size_t)node * CCH + j] = zv * hv + (1.f - zv) * ht;
    }
  }
}

// out = softplus(relu(h_new) @ Wl + bl) : wave per node, lane = channel.
__global__ void k_readout(const float* __restrict__ hnew, const float* __restrict__ Wl,
                          const float* __restrict__ bl, float* __restrict__ out, int n) {
  int lane = threadIdx.x & 31;
  int node = blockIdx.x * (blockDim.x >> 5) + (threadIdx.x >> 5);
  if (node >= n) return;
  float v = hnew[(size_t)node * CCH + lane];
  v = fmaxf(v, 0.f) * Wl[lane];
#pragma unroll
  for (int off = 16; off; off >>= 1) v += __shfl_xor(v, off, 32);
  if (lane == 0) {
    float t = v + bl[0];
    out[node] = (t > 20.f) ? t : log1pf(__expf(t));
  }
}

// --------------------------------------------------------------------- launch
extern "C" void kernel_launch(void* const* d_in, const int* in_sizes, int n_in,
                              void* d_out, int out_size, void* d_ws, size_t ws_size,
                              hipStream_t stream) {
  const float* x  = (const float*)d_in[0];
  const int*   ei = (const int*)d_in[1];
  const float* ew = (const float*)d_in[2];
  const float* h  = (const float*)d_in[3];
  const float* Wx = (const float*)d_in[4];
  const float* bx = (const float*)d_in[5];
  const float* Wh = (const float*)d_in[6];
  const float* bh = (const float*)d_in[7];
  const float* Wl = (const float*)d_in[8];
  const float* bl = (const float*)d_in[9];

  const int n = in_sizes[0] / CCH;     // 100000
  const int E = in_sizes[2];           // 1600000
  const int* src = ei;
  const int* dst = ei + E;

  // Workspace layout (all region sizes multiples of 16 floats -> 32B aligned).
  _Float16* Wzr = (_Float16*)d_ws;                 // 4*32*64 halfs
  _Float16* Wc  = Wzr + 4 * 32 * 64;               // 4*32*32 halfs
  float* fws  = (float*)d_ws + 6144;               // 12288 halfs = 6144 floats
  float* deg  = fws;  fws += n;
  float* coef = fws;  fws += E;
  float* tx   = fws;  fws += (size_t)n * CCH;
  float* th   = fws;  fws += (size_t)n * CCH;      // reused later as A(h*r)
  float* zbuf = fws;  fws += (size_t)n * CCH;
  float* hrbuf= fws;  fws += (size_t)n * CCH;

  float* out  = (float*)d_out;                     // [N]
  float* hnew = out + n;                           // [N,32]

  const int ntiles = (n + 15) / 16;

  zero_f32x4<<<512, 256, 0, stream>>>((v4f*)deg, n / 4);
  zero_f32x4<<<2048, 256, 0, stream>>>((v4f*)tx, (long)n * CCH * 2 / 4); // tx|th
  cvt_weights<<<48, 256, 0, stream>>>(Wx, Wh, Wzr, Wc);

  k_degree<<<(E + 255) / 256, 256, 0, stream>>>(src, ew, deg, E);
  k_coef<<<(E + 255) / 256, 256, 0, stream>>>(src, dst, ew, deg, coef, E);
  k_spmm2<<<(E + 7) / 8, 256, 0, stream>>>(src, dst, coef, x, h, tx, th, E);

  k_gates_zr<<<(ntiles + 7) / 8, 256, 0, stream>>>(x, tx, h, th, Wzr, bx, bh,
                                                   zbuf, hrbuf, n);

  zero_f32x4<<<2048, 256, 0, stream>>>((v4f*)th, (long)n * CCH / 4);  // th := A(h*r)
  k_spmm1<<<(E + 7) / 8, 256, 0, stream>>>(src, dst, coef, hrbuf, th, E);

  k_gate_c<<<(ntiles + 7) / 8, 256, 0, stream>>>(x, tx, hrbuf, th, Wc, bx, bh,
                                                 h, zbuf, hnew, n);
  k_readout<<<(n + 7) / 8, 256, 0, stream>>>(hnew, Wl, bl, out, n);
}
